// Decoder_34170759807510
// MI455X (gfx1250) — matrix operand order repaired and checked by reading, not compiled, not run on hardware
//
#include <hip/hip_runtime.h>
#include <hip/hip_bf16.h>
#include <math.h>

// ---------------------------------------------------------------------------
// Decoder step: attention + 4 zero-state GRU cells + vocab projection.
// Shapes: B=64, SRC=2048, ENC=1024, U=1024, E=512, V=32000, IN0=1536.
// Strategy: bf16 WMMA (f32 accum) for every GEMM; the big attention
// projection is fused with tanh(.)@Ws so the 512MB "c" tensor is never
// materialized; ctx is a streaming batched GEMV (memory bound anyway).
// Each GEMM wave owns 4 N-tiles to amortize the A fragment; next-K panels
// are prefetched with global_prefetch_b8.
// ---------------------------------------------------------------------------

typedef __attribute__((ext_vector_type(16))) __bf16 v16bf;
typedef __attribute__((ext_vector_type(8)))  float  v8f;
typedef __attribute__((ext_vector_type(4)))  float  v4f;

#define B_   64
#define SRC_ 2048
#define ENC_ 1024
#define U_   1024
#define E_   512
#define V_   32000
#define IN0_ 1536

// ----- WMMA operand loaders (fp32 memory -> bf16 lanes, ISA 7.12.2 layout) --

// A: 16x32 tile of row-major A[M x K] starting at (row0, k0).
// lanes 0-15: M=lane, K in {k0..k0+7} u {k0+16..k0+23}
// lanes 16-31: M=lane-16, K in {k0+8..k0+15} u {k0+24..k0+31}
// Contiguous 8-float runs -> two pairs of b128 loads per lane.
__device__ __forceinline__ v16bf load_a_tile(const float* A, int lda,
                                             int row0, int k0, int lane) {
    int row = row0 + (lane & 15);
    int kg  = (lane >> 4) * 8;
    const float* p = A + (size_t)row * lda + k0 + kg;
    v4f x0 = *(const v4f*)(p);
    v4f x1 = *(const v4f*)(p + 4);
    v4f x2 = *(const v4f*)(p + 16);
    v4f x3 = *(const v4f*)(p + 20);
    v16bf a;
#pragma unroll
    for (int i = 0; i < 4; ++i) {
        a[i]      = (__bf16)x0[i];
        a[4 + i]  = (__bf16)x1[i];
        a[8 + i]  = (__bf16)x2[i];
        a[12 + i] = (__bf16)x3[i];
    }
    return a;
}

// B: 32x16 tile of row-major W[K x N] starting at (k0, col0).
// lanes 0-15: N=lane, K in {k0..k0+7} u {k0+16..k0+23}
// lanes 16-31: N=lane-16, K in {k0+8..k0+15} u {k0+24..k0+31}
__device__ __forceinline__ v16bf load_b_tile(const float* W, int ldn,
                                             int k0, int col0, int lane) {
    int col = col0 + (lane & 15);
    int kb  = (lane >> 4) * 8;
    const float* p = W + (size_t)(k0 + kb) * ldn + col;
    v16bf b;
#pragma unroll
    for (int i = 0; i < 16; ++i) b[i] = (__bf16)p[(size_t)(i + (i >= 8 ? 8 : 0)) * ldn];
    return b;
}

__device__ __forceinline__ v8f wmma_bf16(v16bf a, v16bf b, v8f c) {
    return __builtin_amdgcn_wmma_f32_16x16x32_bf16(
        /*neg_a=*/false, a, /*neg_b=*/false, b,
        /*c_mod=*/(short)0, c, /*reuse_a=*/false, /*reuse_b=*/false);
}

// ----- Generic GEMM: C[MxN] = A[MxK] @ W[KxN] (+ bias[N]) ------------------
// One wave per 16x64 output strip (4 WMMA tiles); 8 waves per block.
#define GEMM_NT 4
__global__ void gemm_wmma(const float* __restrict__ A,
                          const float* __restrict__ W,
                          const float* __restrict__ bias,
                          float* __restrict__ C,
                          int M, int N, int K) {
    int ngrp   = (N >> 4) / GEMM_NT;          // groups of 4 N-tiles
    int mtiles = M >> 4;
    int grp    = blockIdx.x * 8 + (threadIdx.x >> 5);
    if (grp >= mtiles * ngrp) return;          // wave-uniform
    int mt   = grp / ngrp;
    int col0 = (grp % ngrp) * (16 * GEMM_NT);
    int lane = threadIdx.x & 31;

    v8f acc[GEMM_NT] = {};
    for (int k0 = 0; k0 < K; k0 += 32) {
        v16bf a = load_a_tile(A, K, mt * 16, k0, lane);
        if (k0 + 32 < K) {
            // prefetch next K panel of A and the streamed B panel
            __builtin_prefetch(A + (size_t)(mt * 16 + (lane & 15)) * K +
                                   (k0 + 32 + (lane >> 4) * 8), 0, 1);
            __builtin_prefetch(W + (size_t)(k0 + 32 + (lane >> 4) * 16) * N +
                                   col0 + (lane & 15), 0, 1);
        }
#pragma unroll
        for (int t = 0; t < GEMM_NT; ++t) {
            v16bf bm = load_b_tile(W, N, k0, col0 + t * 16, lane);
            acc[t] = wmma_bf16(a, bm, acc[t]);
        }
    }
    int row0 = mt * 16 + 8 * (lane >> 4);
    int cl   = lane & 15;
#pragma unroll
    for (int t = 0; t < GEMM_NT; ++t) {
        int col  = col0 + t * 16 + cl;
        float bv = bias ? bias[col] : 0.0f;
#pragma unroll
        for (int v = 0; v < 8; ++v)
            C[(size_t)(row0 + v) * N + col] = acc[t][v] + bv;
    }
}

// ----- Fused attention scores ----------------------------------------------
// s[b,j] = sum_n tanh( h[b,n] + bc[n] + (enc[b,j,:] @ Wc)[n] ) * Ws[n] + bs
// Block: 256 thr (8 waves) -> one 16-row tile of the flattened (b,j) axis.
// Wave w covers N columns [w*128, w*128+128) as 8 WMMA tiles; "c" stays in
// registers, never touches HBM.
__global__ void attn_scores(const float* __restrict__ enc,
                            const float* __restrict__ Wc,
                            const float* __restrict__ bc,
                            const float* __restrict__ hatt,
                            const float* __restrict__ Ws,
                            const float* __restrict__ bs,
                            float* __restrict__ s) {
    __shared__ float srow[16];
    int row0 = blockIdx.x * 16;            // flattened (b*SRC + j); SRC%16==0
    int b    = row0 >> 11;                 // row0 / SRC
    int lane = threadIdx.x & 31;
    int wv   = threadIdx.x >> 5;

    if (threadIdx.x < 16) srow[threadIdx.x] = 0.0f;
    __syncthreads();

    v8f acc[8] = {};
    for (int k0 = 0; k0 < ENC_; k0 += 32) {
        v16bf a = load_a_tile(enc, ENC_, row0, k0, lane);
        if (k0 + 32 < ENC_)
            __builtin_prefetch(enc + (size_t)(row0 + (lane & 15)) * ENC_ +
                                     (k0 + 32 + (lane >> 4) * 8), 0, 1);
#pragma unroll
        for (int t = 0; t < 8; ++t) {
            v16bf bm = load_b_tile(Wc, U_, k0, wv * 128 + t * 16, lane);
            acc[t] = wmma_bf16(a, bm, acc[t]);
        }
    }

    // epilogue: per-row sum of tanh(h + bc + c) * Ws over this wave's cols
    float sums[8] = {0, 0, 0, 0, 0, 0, 0, 0};
    int cl = lane & 15;
#pragma unroll
    for (int t = 0; t < 8; ++t) {
        int col   = wv * 128 + t * 16 + cl;
        float hb  = hatt[(size_t)b * U_ + col] + bc[col];
        float wsv = Ws[col];
#pragma unroll
        for (int v = 0; v < 8; ++v)
            sums[v] += tanhf(acc[t][v] + hb) * wsv;
    }
    int rbase = 8 * (lane >> 4);
#pragma unroll
    for (int v = 0; v < 8; ++v)
        atomicAdd(&srow[rbase + v], sums[v]);     // ds_add_f32
    __syncthreads();
    if (threadIdx.x < 16)
        s[row0 + threadIdx.x] = srow[threadIdx.x] + bs[0];
}

// ----- Softmax over SRC (in place), one block per batch row ----------------
__global__ void softmax_rows(float* __restrict__ s) {
    __shared__ float red[256];
    float* row = s + (size_t)blockIdx.x * SRC_;
    int tid = threadIdx.x;

    float m = -INFINITY;
    for (int j = tid; j < SRC_; j += 256) m = fmaxf(m, row[j]);
    red[tid] = m; __syncthreads();
    for (int st = 128; st > 0; st >>= 1) {
        if (tid < st) red[tid] = fmaxf(red[tid], red[tid + st]);
        __syncthreads();
    }
    m = red[0]; __syncthreads();

    float sum = 0.0f;
    for (int j = tid; j < SRC_; j += 256) {
        float e = expf(row[j] - m);
        row[j] = e; sum += e;
    }
    red[tid] = sum; __syncthreads();
    for (int st = 128; st > 0; st >>= 1) {
        if (tid < st) red[tid] += red[tid + st];
        __syncthreads();
    }
    float inv = 1.0f / red[0];
    for (int j = tid; j < SRC_; j += 256) row[j] *= inv;
}

// ----- ctx[b,n] = sum_j w[b,j] * enc[b,j,n]  (streaming, coalesced in n) ---
__global__ void ctx_reduce(const float* __restrict__ w,
                           const float* __restrict__ enc,
                           float* __restrict__ ctx) {
    int b = blockIdx.x;
    int n = blockIdx.y * 128 + threadIdx.x;
    const float* wb = w + (size_t)b * SRC_;
    const float* eb = enc + ((size_t)b * SRC_) * ENC_ + n;
    float acc = 0.0f;
    for (int j = 0; j < SRC_; ++j)
        acc += wb[j] * eb[(size_t)j * ENC_];
    ctx[(size_t)b * ENC_ + n] = acc;
}

// ----- xin = concat(ctx, emb[x]) -------------------------------------------
__global__ void build_xin(const float* __restrict__ ctx,
                          const float* __restrict__ emb,
                          const int* __restrict__ x,
                          float* __restrict__ xin) {
    int b = blockIdx.x, tid = threadIdx.x;
    float* dst = xin + (size_t)b * IN0_;
    const float* c = ctx + (size_t)b * ENC_;
    for (int i = tid; i < ENC_; i += 256) dst[i] = c[i];
    const float* e = emb + (size_t)x[b] * E_;
    for (int i = tid; i < E_; i += 256) dst[ENC_ + i] = e[i];
}

// ----- GRU gates with h == 0: out = (1 - z) * hc ---------------------------
// xg = xin @ k (no bias yet); hg = b1 (since h==0).
__global__ void gru_gates(const float* __restrict__ xg,
                          const float* __restrict__ bias,  // [2][3U]
                          float* __restrict__ h) {
    int idx = blockIdx.x * 256 + threadIdx.x;     // B*U
    int b = idx >> 10, u = idx & (U_ - 1);
    const float* g  = xg + (size_t)b * (3 * U_);
    const float* b0 = bias;
    const float* b1 = bias + 3 * U_;
    float z  = 1.0f / (1.0f + expf(-(g[u]          + b0[u]          + b1[u])));
    float r  = 1.0f / (1.0f + expf(-(g[u + U_]     + b0[u + U_]     + b1[u + U_])));
    float hc = tanhf(g[u + 2 * U_] + b0[u + 2 * U_] + r * b1[u + 2 * U_]);
    h[idx] = (1.0f - z) * hc;
}

// ----- y = concat(h1f, h1b) ------------------------------------------------
__global__ void build_y(const float* __restrict__ h1f,
                        const float* __restrict__ h1b,
                        float* __restrict__ y) {
    int b = blockIdx.x, tid = threadIdx.x;
    float* dst = y + (size_t)b * (2 * U_);
    const float* f = h1f + (size_t)b * U_;
    const float* bw = h1b + (size_t)b * U_;
    for (int i = tid; i < U_; i += 256) { dst[i] = f[i]; dst[U_ + i] = bw[i]; }
}

// ---------------------------------------------------------------------------

static inline int gemm_blocks(int M, int N) {
    int grps = (M >> 4) * ((N >> 4) / GEMM_NT);
    return (grps + 7) / 8;
}

extern "C" void kernel_launch(void* const* d_in, const int* in_sizes, int n_in,
                              void* d_out, int out_size, void* d_ws, size_t ws_size,
                              hipStream_t stream) {
    (void)in_sizes; (void)n_in; (void)out_size; (void)ws_size;

    const int*   x      = (const int*)  d_in[0];
    const float* state  = (const float*)d_in[1];
    const float* enc    = (const float*)d_in[2];
    const float* att_Wh = (const float*)d_in[3];
    const float* att_bh = (const float*)d_in[4];
    const float* att_Wc = (const float*)d_in[5];
    const float* att_bc = (const float*)d_in[6];
    const float* att_Ws = (const float*)d_in[7];
    const float* att_bs = (const float*)d_in[8];
    const float* emb    = (const float*)d_in[9];
    const float* fw0_k  = (const float*)d_in[10];
    const float* fw0_b  = (const float*)d_in[12];
    const float* fw1_k  = (const float*)d_in[13];
    const float* fw1_b  = (const float*)d_in[15];
    const float* bw0_k  = (const float*)d_in[16];
    const float* bw0_b  = (const float*)d_in[18];
    const float* bw1_k  = (const float*)d_in[19];
    const float* bw1_b  = (const float*)d_in[21];
    const float* fc_W   = (const float*)d_in[22];
    const float* fc_b   = (const float*)d_in[23];

    // workspace layout (floats)
    float* ws   = (float*)d_ws;
    float* hatt = ws;                        // 64*1024
    float* s    = hatt + B_ * U_;            // 64*2048  (scores -> softmax w)
    float* ctx  = s    + B_ * SRC_;          // 64*1024
    float* xin  = ctx  + B_ * ENC_;          // 64*1536
    float* xg   = xin  + B_ * IN0_;          // 64*3072  (reused per GRU cell)
    float* y    = xg   + B_ * 3 * U_;        // 64*2048

    // output layout: out | h0f | h1f | h0b | h1b
    float* out = (float*)d_out;
    float* h0f = out + (size_t)B_ * V_;
    float* h1f = h0f + B_ * U_;
    float* h0b = h1f + B_ * U_;
    float* h1b = h0b + B_ * U_;

    // 1) h = state @ att_Wh + att_bh
    gemm_wmma<<<gemm_blocks(B_, U_), 256, 0, stream>>>(
        state, att_Wh, att_bh, hatt, B_, U_, U_);

    // 2) fused scores: s[b,j] = tanh(h + bc + enc@Wc) @ Ws + bs
    attn_scores<<<(B_ * SRC_) / 16, 256, 0, stream>>>(
        enc, att_Wc, att_bc, hatt, att_Ws, att_bs, s);

    // 3) softmax over SRC (in place -> attention weights)
    softmax_rows<<<B_, 256, 0, stream>>>(s);

    // 4) ctx = sum_j w * enc
    ctx_reduce<<<dim3(B_, ENC_ / 128), 128, 0, stream>>>(s, enc, ctx);

    // 5) xin = concat(ctx, emb[x])
    build_xin<<<B_, 256, 0, stream>>>(ctx, emb, x, xin);

    // 6) forward GRU stack (hidden state is zero -> single GEMM + gates each)
    gemm_wmma<<<gemm_blocks(B_, 3 * U_), 256, 0, stream>>>(
        xin, fw0_k, nullptr, xg, B_, 3 * U_, IN0_);
    gru_gates<<<(B_ * U_) / 256, 256, 0, stream>>>(xg, fw0_b, h0f);

    gemm_wmma<<<gemm_blocks(B_, 3 * U_), 256, 0, stream>>>(
        h0f, fw1_k, nullptr, xg, B_, 3 * U_, U_);
    gru_gates<<<(B_ * U_) / 256, 256, 0, stream>>>(xg, fw1_b, h1f);

    // 7) backward GRU stack
    gemm_wmma<<<gemm_blocks(B_, 3 * U_), 256, 0, stream>>>(
        xin, bw0_k, nullptr, xg, B_, 3 * U_, IN0_);
    gru_gates<<<(B_ * U_) / 256, 256, 0, stream>>>(xg, bw0_b, h0b);

    gemm_wmma<<<gemm_blocks(B_, 3 * U_), 256, 0, stream>>>(
        h0b, bw1_k, nullptr, xg, B_, 3 * U_, U_);
    gru_gates<<<(B_ * U_) / 256, 256, 0, stream>>>(xg, bw1_b, h1b);

    // 8) y = concat(h1f, h1b); out = y @ fc_W + fc_b
    build_y<<<B_, 256, 0, stream>>>(h1f, h1b, y);
    gemm_wmma<<<gemm_blocks(B_, V_), 256, 0, stream>>>(
        y, fc_W, fc_b, out, B_, V_, 2 * U_);
}
